// ParallelSSM_49675591745703
// MI455X (gfx1250) — matrix operand order repaired and checked by reading, not compiled, not run on hardware
//
#include <hip/hip_runtime.h>
#include <hip/hip_bf16.h>

// Problem constants (fixed by the reference harness)
#define DIMV    1024
#define STATEV  64
#define DTRANK  64
#define PROJV   192      // DTRANK + 2*STATEV
#define CHUNKV  64
#define ROWS    8192     // B*L = 2*4096
#define NCHUNKS 128      // B * (L/CHUNK)
#define EPSV    1e-6f

// GEMM blocking: workgroup tile BM x BN (BN template), K-step BK.
#define BM 128
#define BK 32
constexpr int LDA = BK + 16;  // padded LDS row (96B): 16B-aligned, destaggers banks

#if defined(__has_builtin)
#if __has_builtin(__builtin_amdgcn_sched_barrier)
#define SCHED_FENCE() __builtin_amdgcn_sched_barrier(0)
#endif
#endif
#ifndef SCHED_FENCE
#define SCHED_FENCE()
#endif

typedef __attribute__((ext_vector_type(16))) __bf16 v16bf;
typedef __attribute__((ext_vector_type(8)))  __bf16 v8bf;
typedef __attribute__((ext_vector_type(8)))  float  v8f;

// ---------------------------------------------------------------------------
// Split-bf16 helpers: x ~= (float)hi + (float)lo with ~2^-16 relative error.
// ---------------------------------------------------------------------------
__device__ __forceinline__ void split2(float x, __bf16* ph, __bf16* pl) {
  __bf16 h = (__bf16)x;
  *ph = h;
  *pl = (__bf16)(x - (float)h);
}

// ---------------------------------------------------------------------------
// WMMA fragment loaders (wave32, bf16 16x16x32), p = row base in a BK-wide tile.
// A 16x32 (ISA 7.12.2): lane r holds row M=r; g=lane>>4 selects K in
//   [8g,8g+8) (VGPR0-3) + [16+8g,16+8g+8) (VGPR4-7) -> two 16B reads.
// B 32x16: lane r holds col N=r; same K order as A:
//   [8g,8g+8) + [16+8g,16+8g+8) -> two 16B reads.
// ---------------------------------------------------------------------------
__device__ __forceinline__ v16bf frag_a(const __bf16* p, int g) {
  v8bf a0 = *(const v8bf*)(p + 8 * g);
  v8bf a1 = *(const v8bf*)(p + 16 + 8 * g);
  return __builtin_shufflevector(a0, a1, 0,1,2,3,4,5,6,7,8,9,10,11,12,13,14,15);
}
__device__ __forceinline__ v16bf frag_b(const __bf16* p, int g) {
  v8bf b0 = *(const v8bf*)(p + 8 * g);
  v8bf b1 = *(const v8bf*)(p + 16 + 8 * g);
  return __builtin_shufflevector(b0, b1, 0,1,2,3,4,5,6,7,8,9,10,11,12,13,14,15);
}

__device__ __forceinline__ v8f wmma_bf16(v16bf a, v16bf b, v8f c) {
  // (neg_a, A, neg_b, B, c_mod, C, reuse_a, reuse_b)
  return __builtin_amdgcn_wmma_f32_16x16x32_bf16(false, a, false, b, (short)0, c,
                                                 false, false);
}

// ---------------------------------------------------------------------------
// LDS-tiled split-bf16 GEMM core. Block = 256 threads (8 waves, 2x4 grid),
// block tile BM x BN, wave tile 64 x (BN/4), NS = BN/64 N-subtiles per wave.
// Per K-step: stage A/B hi+lo (coalesced 16B chunks) -> LDS, 3 bf16 WMMAs per
// 16x16x32 step (hi*hi + hi*lo + lo*hi). Global prefetch for step k+1 is
// pinned BEFORE the WMMA phase via sched_barrier so L2 latency overlaps math.
// ---------------------------------------------------------------------------
template <int BN>
__device__ __forceinline__ void gemm_block(const __bf16* __restrict__ Ah,
                                           const __bf16* __restrict__ Al,
                                           const __bf16* __restrict__ Wh,
                                           const __bf16* __restrict__ Wl,
                                           int K, int bm, int bn,
                                           v8f (&acc)[4][BN / 64]) {
  constexpr int NS = BN / 64;
  __shared__ __align__(16) __bf16 sAh[BM * LDA];
  __shared__ __align__(16) __bf16 sAl[BM * LDA];
  __shared__ __align__(16) __bf16 sBh[BN * LDA];
  __shared__ __align__(16) __bf16 sBl[BN * LDA];

  const int tid = threadIdx.x;
  const int lane = tid & 31, wv = tid >> 5;
  const int g = lane >> 4, r = lane & 15;
  const int wm = wv >> 2, wn = wv & 3;

#pragma unroll
  for (int ms = 0; ms < 4; ++ms)
#pragma unroll
    for (int ns = 0; ns < NS; ++ns)
#pragma unroll
      for (int i = 0; i < 8; ++i) acc[ms][ns][i] = 0.f;

  // Staging: chunk c -> row c>>2, k-offset (c&3)*8 (16B). A: 2 chunks/thread,
  // B: NS chunks/thread. Coalesced global 16B loads; LDS layout == fragment
  // packing, so no repacking VALU anywhere.
  v8bf rah[2], ral[2], rbh[NS], rbl[NS];

  auto load_stage = [&](int k0) {
#pragma unroll
    for (int j = 0; j < 2; ++j) {
      int c = tid * 2 + j;
      int row = c >> 2, kc = (c & 3) * 8;
      rah[j] = *(const v8bf*)(Ah + (size_t)(bm + row) * K + k0 + kc);
      ral[j] = *(const v8bf*)(Al + (size_t)(bm + row) * K + k0 + kc);
    }
#pragma unroll
    for (int j = 0; j < NS; ++j) {
      int c = tid * NS + j;
      int row = c >> 2, kc = (c & 3) * 8;
      rbh[j] = *(const v8bf*)(Wh + (size_t)(bn + row) * K + k0 + kc);
      rbl[j] = *(const v8bf*)(Wl + (size_t)(bn + row) * K + k0 + kc);
    }
  };
  auto store_stage = [&]() {
#pragma unroll
    for (int j = 0; j < 2; ++j) {
      int c = tid * 2 + j;
      int row = c >> 2, kc = (c & 3) * 8;
      *(v8bf*)(sAh + row * LDA + kc) = rah[j];
      *(v8bf*)(sAl + row * LDA + kc) = ral[j];
    }
#pragma unroll
    for (int j = 0; j < NS; ++j) {
      int c = tid * NS + j;
      int row = c >> 2, kc = (c & 3) * 8;
      *(v8bf*)(sBh + row * LDA + kc) = rbh[j];
      *(v8bf*)(sBl + row * LDA + kc) = rbl[j];
    }
  };

  load_stage(0);
  store_stage();
  const int nk = K / BK;
  for (int kt = 0; kt < nk; ++kt) {
    __syncthreads();  // staged tile visible to all waves
    if (kt + 1 < nk) {
      load_stage((kt + 1) * BK);  // global prefetch...
      SCHED_FENCE();              // ...pinned to issue before the WMMA phase
    }

    v16bf afh[4], afl[4];
#pragma unroll
    for (int ms = 0; ms < 4; ++ms) {
      int arow = wm * 64 + ms * 16 + r;
      afh[ms] = frag_a(sAh + arow * LDA, g);
      afl[ms] = frag_a(sAl + arow * LDA, g);
    }
#pragma unroll
    for (int ns = 0; ns < NS; ++ns) {
      int bcol = wn * (NS * 16) + ns * 16 + r;
      v16bf bfh = frag_b(sBh + bcol * LDA, g);
      v16bf bfl = frag_b(sBl + bcol * LDA, g);
#pragma unroll
      for (int ms = 0; ms < 4; ++ms) {
        acc[ms][ns] = wmma_bf16(afh[ms], bfh, acc[ms][ns]);
        acc[ms][ns] = wmma_bf16(afh[ms], bfl, acc[ms][ns]);
        acc[ms][ns] = wmma_bf16(afl[ms], bfh, acc[ms][ns]);
      }
    }
    __syncthreads();  // all waves done reading this tile
    if (kt + 1 < nk) store_stage();
  }
}

// ---------------------------------------------------------------------------
// Kernel 1: generic fp32 -> (hi,lo) bf16 plane split (weights).
// ---------------------------------------------------------------------------
__global__ void k_split(const float* __restrict__ src, __bf16* __restrict__ h,
                        __bf16* __restrict__ l, int n) {
  int i = blockIdx.x * 256 + threadIdx.x;
  if (i < n) split2(src[i], h + i, l + i);
}

// ---------------------------------------------------------------------------
// Kernel 2: RMSNorm, emits xn as hi/lo bf16 planes (GEMM-A format).
// ---------------------------------------------------------------------------
__global__ void k_rmsnorm(const float* __restrict__ x, const float* __restrict__ w,
                          __bf16* __restrict__ xh, __bf16* __restrict__ xl) {
  int row = blockIdx.x;
  int tid = threadIdx.x;
  const float* xr = x + (size_t)row * DIMV;
  float v[4];
  float s = 0.f;
#pragma unroll
  for (int i = 0; i < 4; ++i) {
    v[i] = xr[tid + 256 * i];
    s += v[i] * v[i];
  }
#pragma unroll
  for (int o = 16; o > 0; o >>= 1) s += __shfl_xor(s, o, 32);
  __shared__ float red[8];
  if ((tid & 31) == 0) red[tid >> 5] = s;
  __syncthreads();
  if (tid == 0) {
    float t = 0.f;
#pragma unroll
    for (int i = 0; i < 8; ++i) t += red[i];
    red[0] = rsqrtf(t * (1.0f / DIMV) + EPSV);
  }
  __syncthreads();
  float scale = red[0];
#pragma unroll
  for (int i = 0; i < 4; ++i) {
    int c = tid + 256 * i;
    float xn = v[i] * scale * w[c];
    split2(xn, xh + (size_t)row * DIMV + c, xl + (size_t)row * DIMV + c);
  }
}

// ---------------------------------------------------------------------------
// Kernel 3: proj = xn @ x_proj_w^T   (8192 x 192 x 1024), BN=64 (192 = 3*64)
// cols 0..63   -> delta_lo as hi/lo bf16 planes; cols 64..191 -> B/C fp32.
// ---------------------------------------------------------------------------
__global__ void __launch_bounds__(256) k_proj(
    const __bf16* __restrict__ xh, const __bf16* __restrict__ xl,
    const __bf16* __restrict__ wh, const __bf16* __restrict__ wl,
    float* __restrict__ bc, __bf16* __restrict__ dlh, __bf16* __restrict__ dll) {
  v8f acc[4][1];
  const int bm = blockIdx.x * BM, bn = blockIdx.y * 64;
  gemm_block<64>(xh, xl, wh, wl, DIMV, bm, bn, acc);
  const int lane = threadIdx.x & 31, wv = threadIdx.x >> 5;
  const int g = lane >> 4, r = lane & 15;
  const int wm = wv >> 2, wn = wv & 3;
  const int col = bn + wn * 16 + r;
#pragma unroll
  for (int ms = 0; ms < 4; ++ms)
#pragma unroll
    for (int i = 0; i < 8; ++i) {
      int row = bm + wm * 64 + ms * 16 + 8 * g + i;
      float val = acc[ms][0][i];
      if (col < DTRANK) {
        split2(val, dlh + (size_t)row * DTRANK + col,
               dll + (size_t)row * DTRANK + col);
      } else {
        bc[(size_t)row * 128 + (col - DTRANK)] = val;
      }
    }
}

// ---------------------------------------------------------------------------
// Kernel 4: delta = softplus(delta_lo @ dt_proj_w^T + b) (8192 x 1024 x 64)
// ---------------------------------------------------------------------------
__global__ void __launch_bounds__(256) k_dt(
    const __bf16* __restrict__ ah, const __bf16* __restrict__ al,
    const __bf16* __restrict__ wh, const __bf16* __restrict__ wl,
    const float* __restrict__ bias, float* __restrict__ delta) {
  v8f acc[4][2];
  const int bm = blockIdx.x * BM, bn = blockIdx.y * 128;
  gemm_block<128>(ah, al, wh, wl, DTRANK, bm, bn, acc);
  const int lane = threadIdx.x & 31, wv = threadIdx.x >> 5;
  const int g = lane >> 4, r = lane & 15;
  const int wm = wv >> 2, wn = wv & 3;
#pragma unroll
  for (int ns = 0; ns < 2; ++ns) {
    int col = bn + wn * 32 + ns * 16 + r;
    float b = bias[col];
#pragma unroll
    for (int ms = 0; ms < 4; ++ms)
#pragma unroll
      for (int i = 0; i < 8; ++i) {
        int row = bm + wm * 64 + ms * 16 + 8 * g + i;
        float z = acc[ms][ns][i] + b;
        float sp = (z > 20.f) ? z : log1pf(__expf(z));
        delta[(size_t)row * DIMV + col] = sp;
      }
  }
}

// ---------------------------------------------------------------------------
// Kernel 5: chunk-local selective scan. One thread per channel d, state h[64]
// and decay A[64] in VGPRs; B/C for the whole chunk staged in LDS (32 KB).
// Fuses y + x*D skip and emits hi/lo bf16 planes for the output GEMM.
// ---------------------------------------------------------------------------
__global__ void __launch_bounds__(256) k_scan(
    const float* __restrict__ bc, const float* __restrict__ delta,
    const __bf16* __restrict__ xh, const __bf16* __restrict__ xl,
    const float* __restrict__ x, const float* __restrict__ A_log,
    const float* __restrict__ Dv, __bf16* __restrict__ yh,
    __bf16* __restrict__ yl) {
  int m = blockIdx.x;                      // chunk 0..127
  int d = blockIdx.y * 256 + threadIdx.x;  // channel 0..1023
  int tid = threadIdx.x;

  __shared__ float Bs[CHUNKV][STATEV];
  __shared__ float Cs[CHUNKV][STATEV];
  for (int idx = tid; idx < CHUNKV * 128; idx += 256) {
    int t = idx >> 7, j = idx & 127;
    float v = bc[((size_t)(m * CHUNKV + t)) * 128 + j];
    if (j < STATEV) Bs[t][j] = v;
    else            Cs[t][j - STATEV] = v;
  }
  __syncthreads();

  float A[STATEV];
#pragma unroll
  for (int n = 0; n < STATEV; ++n) A[n] = -__expf(A_log[(size_t)d * STATEV + n]);
  float h[STATEV];
#pragma unroll
  for (int n = 0; n < STATEV; ++n) h[n] = 0.f;
  float dskip = Dv[d];

  for (int t = 0; t < CHUNKV; ++t) {
    size_t off = (size_t)(m * CHUNKV + t) * DIMV + d;
    float dt  = delta[off];
    float xv  = (float)xh[off] + (float)xl[off];
    float dtx = dt * xv;
    float y = 0.f;
#pragma unroll
    for (int n = 0; n < STATEV; ++n) {
      float dA = __expf(dt * A[n]);  // v_exp_f32 (TRANS, co-executes with VALU)
      h[n] = fmaf(dA, h[n], dtx * Bs[t][n]);
      y = fmaf(h[n], Cs[t][n], y);
    }
    float ys = y + x[off] * dskip;
    split2(ys, yh + off, yl + off);
  }
}

// ---------------------------------------------------------------------------
// Kernel 6: out = y_skip @ out_proj_w^T   (8192 x 1024 x 1024), BN=128
// ---------------------------------------------------------------------------
__global__ void __launch_bounds__(256) k_out(
    const __bf16* __restrict__ ah, const __bf16* __restrict__ al,
    const __bf16* __restrict__ wh, const __bf16* __restrict__ wl,
    float* __restrict__ out) {
  v8f acc[4][2];
  const int bm = blockIdx.x * BM, bn = blockIdx.y * 128;
  gemm_block<128>(ah, al, wh, wl, DIMV, bm, bn, acc);
  const int lane = threadIdx.x & 31, wv = threadIdx.x >> 5;
  const int g = lane >> 4, r = lane & 15;
  const int wm = wv >> 2, wn = wv & 3;
#pragma unroll
  for (int ns = 0; ns < 2; ++ns) {
    int col = bn + wn * 32 + ns * 16 + r;
#pragma unroll
    for (int ms = 0; ms < 4; ++ms)
#pragma unroll
      for (int i = 0; i < 8; ++i) {
        int row = bm + wm * 64 + ms * 16 + 8 * g + i;
        out[(size_t)row * DIMV + col] = acc[ms][ns][i];
      }
  }
}

// ---------------------------------------------------------------------------
// Host launcher
// ---------------------------------------------------------------------------
static inline char* carve(char*& p, size_t bytes) {
  char* r = p;
  p += (bytes + 255) & ~(size_t)255;
  return r;
}

extern "C" void kernel_launch(void* const* d_in, const int* in_sizes, int n_in,
                              void* d_out, int out_size, void* d_ws, size_t ws_size,
                              hipStream_t stream) {
  const float* x         = (const float*)d_in[0];  // [2,4096,1024]
  const float* norm_w    = (const float*)d_in[1];  // [1024]
  const float* x_proj_w  = (const float*)d_in[2];  // [192,1024]
  const float* dt_proj_w = (const float*)d_in[3];  // [1024,64]
  const float* dt_proj_b = (const float*)d_in[4];  // [1024]
  const float* A_log     = (const float*)d_in[5];  // [1024,64]
  const float* Dv        = (const float*)d_in[6];  // [1024]
  const float* out_w     = (const float*)d_in[7];  // [1024,1024]
  float* out = (float*)d_out;

  char* p = (char*)d_ws;
  __bf16* xh   = (__bf16*)carve(p, (size_t)ROWS * DIMV * 2);
  __bf16* xl   = (__bf16*)carve(p, (size_t)ROWS * DIMV * 2);
  float*  bcw  = (float*) carve(p, (size_t)ROWS * 128 * 4);
  __bf16* dlh  = (__bf16*)carve(p, (size_t)ROWS * DTRANK * 2);
  __bf16* dll  = (__bf16*)carve(p, (size_t)ROWS * DTRANK * 2);
  float*  dlt  = (float*) carve(p, (size_t)ROWS * DIMV * 4);
  __bf16* yh   = (__bf16*)carve(p, (size_t)ROWS * DIMV * 2);
  __bf16* yl   = (__bf16*)carve(p, (size_t)ROWS * DIMV * 2);
  __bf16* xwh  = (__bf16*)carve(p, (size_t)PROJV * DIMV * 2);
  __bf16* xwl  = (__bf16*)carve(p, (size_t)PROJV * DIMV * 2);
  __bf16* dtwh = (__bf16*)carve(p, (size_t)DIMV * DTRANK * 2);
  __bf16* dtwl = (__bf16*)carve(p, (size_t)DIMV * DTRANK * 2);
  __bf16* owh  = (__bf16*)carve(p, (size_t)DIMV * DIMV * 2);
  __bf16* owl  = (__bf16*)carve(p, (size_t)DIMV * DIMV * 2);

  // Weight splits (hi/lo bf16 planes)
  {
    int n1 = PROJV * DIMV, n2 = DIMV * DTRANK, n3 = DIMV * DIMV;
    k_split<<<(n1 + 255) / 256, 256, 0, stream>>>(x_proj_w, xwh, xwl, n1);
    k_split<<<(n2 + 255) / 256, 256, 0, stream>>>(dt_proj_w, dtwh, dtwl, n2);
    k_split<<<(n3 + 255) / 256, 256, 0, stream>>>(out_w, owh, owl, n3);
  }

  k_rmsnorm<<<ROWS, 256, 0, stream>>>(x, norm_w, xh, xl);

  k_proj<<<dim3(ROWS / BM, PROJV / 64), 256, 0, stream>>>(
      xh, xl, xwh, xwl, bcw, dlh, dll);

  k_dt<<<dim3(ROWS / BM, DIMV / 128), 256, 0, stream>>>(
      dlh, dll, dtwh, dtwl, dt_proj_b, dlt);

  k_scan<<<dim3(NCHUNKS, DIMV / 256), 256, 0, stream>>>(
      bcw, dlt, xh, xl, x, A_log, Dv, yh, yl);

  k_out<<<dim3(ROWS / BM, DIMV / 128), 256, 0, stream>>>(
      yh, yl, owh, owl, out);
}
